// MultiHeadSelfAttention_41661182771779
// MI455X (gfx1250) — matrix operand order repaired and checked by reading, not compiled, not run on hardware
//
#include <hip/hip_runtime.h>
#include <hip/hip_bf16.h>

// ---- problem constants (B=2, M=2048, HID=1024, 16 heads x 64) ----
#define BATCH 2
#define SEQ   2048
#define HIDN  1024
#define NHEAD 16
#define HD    64
#define MTOT  (BATCH * SEQ)   // 4096

typedef __bf16 bf16;
typedef __attribute__((ext_vector_type(16))) __bf16 v16bf;
typedef __attribute__((ext_vector_type(8)))  __bf16 v8bf;
typedef __attribute__((ext_vector_type(8)))  float  v8f;

// ---------------- WMMA tile load/store helpers (wave32 layouts) -------------
// A 16x32 bf16 tile: lane<16 holds row (lane), K = {k0..k0+7, k0+16..k0+23};
// lane>=16 holds row (lane-16), K shifted by 8.  (ISA 05_wmma §7.12.2)
__device__ __forceinline__ v16bf load_a_tile(const bf16* __restrict__ base,
                                             int ld, int row0, int k0, int lane) {
  const int r  = row0 + (lane & 15);
  const int kk = k0 + ((lane >> 4) << 3);
  union { v16bf v; v8bf h[2]; } u;
  u.h[0] = *(const v8bf*)(base + (size_t)r * ld + kk);
  u.h[1] = *(const v8bf*)(base + (size_t)r * ld + kk + 16);
  return u.v;
}

// B 32x16 bf16 tile: lane<16 holds column (lane), K = {k0..k0+7, k0+16..k0+23};
// lane>=16 holds column (lane-16), K shifted by 8.  Source rows = B columns
// (i.e. W stored [n][k] row-major => two 16B loads).
__device__ __forceinline__ v16bf load_b_tile(const bf16* __restrict__ base,
                                             int ld, int n0, int k0, int lane) {
  const int n  = n0 + (lane & 15);
  const int kk = k0 + ((lane >> 4) << 3);
  union { v16bf v; v8bf h[2]; } u;
  u.h[0] = *(const v8bf*)(base + (size_t)n * ld + kk);
  u.h[1] = *(const v8bf*)(base + (size_t)n * ld + kk + 16);
  return u.v;
}

// D/C 16x16 f32: vgpr j -> (M = (lane>>4)*8 + j, N = lane&15)
__device__ __forceinline__ void store_d_f32(float* __restrict__ C, int ld,
                                            int m0, int n0, int lane, v8f acc) {
  const int n  = n0 + (lane & 15);
  const int mm = m0 + ((lane >> 4) << 3);
#pragma unroll
  for (int j = 0; j < 8; ++j) C[(size_t)(mm + j) * ld + n] = acc[j];
}

__device__ __forceinline__ void store_d_bf16(bf16* __restrict__ C, int ld,
                                             int m0, int n0, int lane, v8f acc) {
  const int n  = n0 + (lane & 15);
  const int mm = m0 + ((lane >> 4) << 3);
#pragma unroll
  for (int j = 0; j < 8; ++j) C[(size_t)(mm + j) * ld + n] = (bf16)acc[j];
}

#define WMMA_BF16(a, b, c) \
  __builtin_amdgcn_wmma_f32_16x16x32_bf16(false, (a), false, (b), (short)0, (c), false, false)

// 32m x 64n register-blocked strip: 8 WMMAs per k-step off 4 A + 8 B b128-loads
// (1.5 loads/WMMA vs 2.5 for 16x64 blocking).
__device__ __forceinline__ void gemm_strip_32x64(const bf16* __restrict__ A, int lda,
                                                 const bf16* __restrict__ Wb, int ldb,
                                                 int m0, int n0, int kLen, int lane,
                                                 v8f acc[2][4]) {
  for (int k0 = 0; k0 < kLen; k0 += 32) {
    v16bf a0 = load_a_tile(A, lda, m0,      k0, lane);
    v16bf a1 = load_a_tile(A, lda, m0 + 16, k0, lane);
#pragma unroll
    for (int t = 0; t < 4; ++t) {
      v16bf b = load_b_tile(Wb, ldb, n0 + t * 16, k0, lane);
      acc[0][t] = WMMA_BF16(a0, b, acc[0][t]);
      acc[1][t] = WMMA_BF16(a1, b, acc[1][t]);
    }
  }
}

// ---------------------------- kernels ---------------------------------------
__global__ void k_cvt_bf16(const float* __restrict__ s, bf16* __restrict__ d, int n) {
  int i = blockIdx.x * blockDim.x + threadIdx.x;
  if (i < n) d[i] = (bf16)s[i];
}

// C[MTOT x 1024] = A[MTOT x 1024] @ W[1024 x 1024]^T  (bf16 in, f32 acc)
// transposed==0: store bf16 row-major into Crm[MTOT x 1024]
// transposed==1: store per-(batch,feature) transposed: Ctr[(b*HIDN + n) * SEQ + m]
// grid = (HIDN/64, MTOT/256), block = 256 (8 waves; each wave -> 32m x 64n strip)
__global__ void k_gemm_proj(const bf16* __restrict__ A, const bf16* __restrict__ W,
                            bf16* __restrict__ Crm, bf16* __restrict__ Ctr,
                            int transposed) {
  const int lane = threadIdx.x & 31;
  const int wave = threadIdx.x >> 5;
  const int m0 = blockIdx.y * 256 + wave * 32;
  const int n0 = blockIdx.x * 64;
  v8f acc[2][4] = {};
  gemm_strip_32x64(A, HIDN, W, HIDN, m0, n0, HIDN, lane, acc);
  if (!transposed) {
#pragma unroll
    for (int rt = 0; rt < 2; ++rt)
#pragma unroll
      for (int t = 0; t < 4; ++t)
        store_d_bf16(Crm, HIDN, m0 + rt * 16, n0 + t * 16, lane, acc[rt][t]);
  } else {
    const int bb = m0 / SEQ;                       // tiles never straddle batches
#pragma unroll
    for (int rt = 0; rt < 2; ++rt) {
      const int ml = (m0 % SEQ) + rt * 16 + ((lane >> 4) << 3);
#pragma unroll
      for (int t = 0; t < 4; ++t) {
        const int n = n0 + t * 16 + (lane & 15);
        v8bf p;
#pragma unroll
        for (int j = 0; j < 8; ++j) p[j] = (bf16)acc[rt][t][j];
        *(v8bf*)(Ctr + (size_t)(bb * HIDN + n) * SEQ + ml) = p;  // one 16B store
      }
    }
  }
}

// T^T[d2][d1] = sum_m V[m][d2] * K[m][d1], per (batch,head).
// Vt/Kt stored transposed [bh*64 + d][SEQ] -> both operands contiguous.
// grid = 32 (bh), block = 128 (4 waves; wave w -> d2 rows w*16..+15, all 64 d1)
__global__ void k_kv_outer(const bf16* __restrict__ Kt, const bf16* __restrict__ Vt,
                           bf16* __restrict__ Tt) {
  const int lane = threadIdx.x & 31;
  const int wave = threadIdx.x >> 5;
  const int bh = blockIdx.x;
  const bf16* Kh = Kt + (size_t)bh * HD * SEQ;
  const bf16* Vh = Vt + (size_t)bh * HD * SEQ;
  bf16* Th = Tt + (size_t)bh * HD * HD;
  const int r0 = wave * 16;
  v8f acc[4] = {};
  for (int m0 = 0; m0 < SEQ; m0 += 32) {
    v16bf a = load_a_tile(Vh, SEQ, r0, m0, lane);
#pragma unroll
    for (int t = 0; t < 4; ++t) {
      v16bf b = load_b_tile(Kh, SEQ, t * 16, m0, lane);
      acc[t] = WMMA_BF16(a, b, acc[t]);
    }
  }
#pragma unroll
  for (int t = 0; t < 4; ++t) store_d_bf16(Th, HD, r0, t * 16, lane, acc[t]);
}

// O[m][d2] = sum_d1 Q[m][d1] * T[d1][d2]; B operand reads T^T row-major.
// grid = (32 bh, SEQ/256), block = 256 (8 waves; 32m x 64n strips)
__global__ void k_attn_apply(const bf16* __restrict__ Q, const bf16* __restrict__ Tt,
                             bf16* __restrict__ O) {
  const int lane = threadIdx.x & 31;
  const int wave = threadIdx.x >> 5;
  const int bh = blockIdx.x;
  const int bb = bh >> 4, hh = bh & 15;
  const int m0 = blockIdx.y * 256 + wave * 32;
  const bf16* Qh = Q + (size_t)bb * SEQ * HIDN + hh * HD;
  const bf16* Th = Tt + (size_t)bh * HD * HD;
  bf16* Oh = O + (size_t)bb * SEQ * HIDN + hh * HD;
  v8f acc[2][4] = {};
  gemm_strip_32x64(Qh, HIDN, Th, HD, m0, 0, HD, lane, acc);
#pragma unroll
  for (int rt = 0; rt < 2; ++rt)
#pragma unroll
    for (int t = 0; t < 4; ++t)
      store_d_bf16(Oh, HIDN, m0 + rt * 16, t * 16, lane, acc[rt][t]);
}

// out[MTOT x 1024] (f32) = O @ Wo^T;  grid = (16, 16), block = 256
__global__ void k_gemm_out(const bf16* __restrict__ A, const bf16* __restrict__ W,
                           float* __restrict__ out) {
  const int lane = threadIdx.x & 31;
  const int wave = threadIdx.x >> 5;
  const int m0 = blockIdx.y * 256 + wave * 32;
  const int n0 = blockIdx.x * 64;
  v8f acc[2][4] = {};
  gemm_strip_32x64(A, HIDN, W, HIDN, m0, n0, HIDN, lane, acc);
#pragma unroll
  for (int rt = 0; rt < 2; ++rt)
#pragma unroll
    for (int t = 0; t < 4; ++t)
      store_d_f32(out, HIDN, m0 + rt * 16, n0 + t * 16, lane, acc[rt][t]);
}

// ---------------------------- launcher ---------------------------------------
extern "C" void kernel_launch(void* const* d_in, const int* in_sizes, int n_in,
                              void* d_out, int out_size, void* d_ws, size_t ws_size,
                              hipStream_t stream) {
  const float* h  = (const float*)d_in[0];
  const float* Wq = (const float*)d_in[1];
  const float* Wk = (const float*)d_in[2];
  const float* Wv = (const float*)d_in[3];
  // d_in[4] = Wspan: dead code in the reference output -> skipped entirely
  const float* Wo = (const float*)d_in[5];
  float* out = (float*)d_out;

  // workspace carve-up (bf16 elements); total ~48.3 MB
  bf16* p    = (bf16*)d_ws;
  bf16* hbf  = p; p += (size_t)MTOT * HIDN;
  bf16* Wqb  = p; p += (size_t)HIDN * HIDN;
  bf16* Wkb  = p; p += (size_t)HIDN * HIDN;
  bf16* Wvb  = p; p += (size_t)HIDN * HIDN;
  bf16* Wob  = p; p += (size_t)HIDN * HIDN;
  bf16* Qb   = p; p += (size_t)MTOT * HIDN;        // [b*SEQ+m][HIDN] row-major
  bf16* Ktb  = p; p += (size_t)MTOT * HIDN;        // [(b*HIDN+n)][SEQ] transposed
  bf16* Vtb  = p; p += (size_t)MTOT * HIDN;        // transposed
  bf16* Ob   = p; p += (size_t)MTOT * HIDN;        // row-major
  bf16* Ttb  = p; p += (size_t)BATCH * NHEAD * HD * HD;  // T^T per head, 64x64

  const int nH = MTOT * HIDN;     // 4194304
  const int nW = HIDN * HIDN;     // 1048576
  k_cvt_bf16<<<(nH + 255) / 256, 256, 0, stream>>>(h,  hbf, nH);
  k_cvt_bf16<<<(nW + 255) / 256, 256, 0, stream>>>(Wq, Wqb, nW);
  k_cvt_bf16<<<(nW + 255) / 256, 256, 0, stream>>>(Wk, Wkb, nW);
  k_cvt_bf16<<<(nW + 255) / 256, 256, 0, stream>>>(Wv, Wvb, nW);
  k_cvt_bf16<<<(nW + 255) / 256, 256, 0, stream>>>(Wo, Wob, nW);

  dim3 gGemm(HIDN / 64, MTOT / 256);   // (16, 16)
  k_gemm_proj<<<gGemm, 256, 0, stream>>>(hbf, Wqb, Qb,  Qb,  0);  // Q row-major
  k_gemm_proj<<<gGemm, 256, 0, stream>>>(hbf, Wkb, Ktb, Ktb, 1);  // K transposed
  k_gemm_proj<<<gGemm, 256, 0, stream>>>(hbf, Wvb, Vtb, Vtb, 1);  // V transposed

  k_kv_outer<<<BATCH * NHEAD, 128, 0, stream>>>(Ktb, Vtb, Ttb);   // T^T = V^T K

  k_attn_apply<<<dim3(BATCH * NHEAD, SEQ / 256), 256, 0, stream>>>(Qb, Ttb, Ob);

  k_gemm_out<<<gGemm, 256, 0, stream>>>(Ob, Wob, out);

  (void)in_sizes; (void)n_in; (void)out_size; (void)ws_size;
}
